// SeedGNN_GMWM_32847909880050
// MI455X (gfx1250) — matrix operand order repaired and checked by reading, not compiled, not run on hardware
//
#include <hip/hip_runtime.h>
#include <stdint.h>

typedef __bf16 bf16_t;
typedef __attribute__((ext_vector_type(16))) __bf16 v16bf;
typedef __attribute__((ext_vector_type(8)))  __bf16 v8bf;
typedef __attribute__((ext_vector_type(8)))  float  v8f;

#define NDIM 1024
#define NN   (1024u * 1024u)
#define HID  16
#define NSEEDS 50
#define APITCH 40   // LDS row pitch (bf16 elems) for 32-K tiles: 80B -> conflict-free b128 reads
#define BPITCH 40

// ============================================================================
// GEMM: C(bf16)[1024x1024] = A(bf16, row-major) x B, B supplied pre-transposed
// as Bt[n][k]. Double-buffered async global->LDS staging (ASYNCcnt) feeding
// v_wmma_f32_16x16x32_bf16 (fp32 accumulate). Block = 8 wave32 waves,
// block tile 128(M) x 64(N), per-wave 32x32 (2x2 fragments), K-step 32.
// ============================================================================
__global__ __launch_bounds__(256) void gemm_bf16_nt(const bf16_t* __restrict__ A,
                                                    const bf16_t* __restrict__ Bt,
                                                    bf16_t* __restrict__ C) {
  __shared__ bf16_t Asm[2][128 * APITCH];
  __shared__ bf16_t Bsm[2][64 * BPITCH];

  const int t    = threadIdx.x;
  const int lane = t & 31;
  const int wave = t >> 5;
  const int wm   = wave & 3;            // 4 waves along M
  const int wn   = wave >> 2;           // 2 waves along N
  const int mBlock = blockIdx.y * 128;
  const int nBlock = blockIdx.x * 64;
  const int hi  = lane >> 4;            // lane group (0: lanes 0-15, 1: lanes 16-31)
  const int l15 = lane & 15;

  // Per-thread async-copy chunk coordinates (3 x 16B chunks per K-step):
  //   A tile 128x32 bf16 = 512 x 16B chunks -> 2/thread; B tile 64x32 = 256 -> 1/thread.
  const int aRow0 = t >> 2;             // rows 0..63
  const int aRow1 = (t + 256) >> 2;     // rows 64..127
  const int aKc   = (t & 3) * 8;        // k-chunk within tile (elements)
  const int bRow  = t >> 2;             // B "rows" are output columns n: 0..63

  auto issueTile = [&](int kb, int buf) {
    {
      unsigned lds = (unsigned)(uintptr_t)&Asm[buf][aRow0 * APITCH + aKc];
      const bf16_t* g = A + (size_t)(mBlock + aRow0) * NDIM + kb + aKc;
      asm volatile("global_load_async_to_lds_b128 %0, %1, off" :: "v"(lds), "v"(g) : "memory");
    }
    {
      unsigned lds = (unsigned)(uintptr_t)&Asm[buf][aRow1 * APITCH + aKc];
      const bf16_t* g = A + (size_t)(mBlock + aRow1) * NDIM + kb + aKc;
      asm volatile("global_load_async_to_lds_b128 %0, %1, off" :: "v"(lds), "v"(g) : "memory");
    }
    {
      unsigned lds = (unsigned)(uintptr_t)&Bsm[buf][bRow * BPITCH + aKc];
      const bf16_t* g = Bt + (size_t)(nBlock + bRow) * NDIM + kb + aKc;
      asm volatile("global_load_async_to_lds_b128 %0, %1, off" :: "v"(lds), "v"(g) : "memory");
    }
  };

  v8f c00 = {}, c01 = {}, c10 = {}, c11 = {};

  int buf = 0;
  issueTile(0, 0);

  for (int kb = 0; kb < NDIM; kb += 32) {
    const bool hasNext = (kb + 32) < NDIM;
    if (hasNext) issueTile(kb + 32, buf ^ 1);
    // Previous 3 async loads complete in order; <=3 outstanding means current
    // tile's data has landed in LDS.
    if (hasNext) asm volatile("s_wait_asynccnt 0x3" ::: "memory");
    else         asm volatile("s_wait_asynccnt 0x0" ::: "memory");
    __syncthreads();

    const bf16_t* As = &Asm[buf][(wm * 32) * APITCH];
    const bf16_t* Bs = &Bsm[buf][(wn * 32) * BPITCH];

    // A fragment (16-bit A 16x32): lane row = lane&15; lanes 0-15 hold K=0..7,16..23;
    // lanes 16-31 hold K=8..15,24..31.
    const int ka = hi * 8;
    v8bf a0lo = *(const v8bf*)(As + l15 * APITCH + ka);
    v8bf a0hi = *(const v8bf*)(As + l15 * APITCH + ka + 16);
    v8bf a1lo = *(const v8bf*)(As + (16 + l15) * APITCH + ka);
    v8bf a1hi = *(const v8bf*)(As + (16 + l15) * APITCH + ka + 16);
    v16bf a0 = __builtin_shufflevector(a0lo, a0hi, 0,1,2,3,4,5,6,7,8,9,10,11,12,13,14,15);
    v16bf a1 = __builtin_shufflevector(a1lo, a1hi, 0,1,2,3,4,5,6,7,8,9,10,11,12,13,14,15);
    // B fragment (16-bit B 32x16): lane col = lane&15; lanes 0-15 hold K=0..7,16..23;
    // lanes 16-31 hold K=8..15,24..31.
    const int kc = hi * 8;
    v8bf b0lo = *(const v8bf*)(Bs + l15 * BPITCH + kc);
    v8bf b0hi = *(const v8bf*)(Bs + l15 * BPITCH + kc + 16);
    v8bf b1lo = *(const v8bf*)(Bs + (16 + l15) * BPITCH + kc);
    v8bf b1hi = *(const v8bf*)(Bs + (16 + l15) * BPITCH + kc + 16);
    v16bf b0 = __builtin_shufflevector(b0lo, b0hi, 0,1,2,3,4,5,6,7,8,9,10,11,12,13,14,15);
    v16bf b1 = __builtin_shufflevector(b1lo, b1hi, 0,1,2,3,4,5,6,7,8,9,10,11,12,13,14,15);

    c00 = __builtin_amdgcn_wmma_f32_16x16x32_bf16(false, a0, false, b0, (short)0, c00, false, false);
    c01 = __builtin_amdgcn_wmma_f32_16x16x32_bf16(false, a0, false, b1, (short)0, c01, false, false);
    c10 = __builtin_amdgcn_wmma_f32_16x16x32_bf16(false, a1, false, b0, (short)0, c10, false, false);
    c11 = __builtin_amdgcn_wmma_f32_16x16x32_bf16(false, a1, false, b1, (short)0, c11, false, false);

    __syncthreads();
    buf ^= 1;
  }

  // C layout: VGPR r holds row (r + 8*hi), column = lane&15.
  const int mBase = mBlock + wm * 32;
  const int nBase = nBlock + wn * 32;
  const int cr = hi * 8;
  #pragma unroll
  for (int r = 0; r < 8; ++r) {
    C[(size_t)(mBase + cr + r)      * NDIM + nBase + l15]      = (bf16_t)c00[r];
    C[(size_t)(mBase + cr + r)      * NDIM + nBase + 16 + l15] = (bf16_t)c01[r];
    C[(size_t)(mBase + 16 + cr + r) * NDIM + nBase + l15]      = (bf16_t)c10[r];
    C[(size_t)(mBase + 16 + cr + r) * NDIM + nBase + 16 + l15] = (bf16_t)c11[r];
  }
}

// ============================================================================
// Helpers / pointwise kernels
// ============================================================================
__global__ void cvt_f32_to_bf16(const float* __restrict__ in, bf16_t* __restrict__ out, unsigned n) {
  for (unsigned i = blockIdx.x * blockDim.x + threadIdx.x; i < n; i += gridDim.x * blockDim.x)
    out[i] = (bf16_t)in[i];
}

__global__ void zero_u32(uint32_t* __restrict__ p, unsigned n) {
  for (unsigned i = blockIdx.x * blockDim.x + threadIdx.x; i < n; i += gridDim.x * blockDim.x)
    p[i] = 0u;
}

// out[c][k] = (bf16) in[k][c]  — LDS-tiled transpose+convert (G2 -> G2T)
__global__ __launch_bounds__(256) void transpose_f32_to_bf16(const float* __restrict__ in,
                                                             bf16_t* __restrict__ out) {
  __shared__ bf16_t tile[32][33];
  const int bx = blockIdx.x * 32;
  const int by = blockIdx.y * 32;
  const int tx = threadIdx.x & 31;
  const int ty = threadIdx.x >> 5;
  for (int r = ty; r < 32; r += 8)
    tile[r][tx] = (bf16_t)in[(size_t)(by + r) * NDIM + bx + tx];
  __syncthreads();
  for (int r = ty; r < 32; r += 8)
    out[(size_t)(bx + r) * NDIM + by + tx] = tile[tx][r];
}

// ST0[s1[m]][s0[m]] = 1  (S stored transposed: ST[k2][j] = S[j][k2])
__global__ void set_seed_plane(bf16_t* __restrict__ ST0, const int* __restrict__ seeds) {
  int t = threadIdx.x;
  if (t < NSEEDS) {
    int j = seeds[t], k2 = seeds[NSEEDS + t];
    ST0[(size_t)k2 * NDIM + j] = (bf16_t)1.0f;
  }
}

// Match[i,c] = ro_b + sum_h H_h[i,c] * ro_W[h]
__global__ void readout_kernel(const bf16_t* __restrict__ H, const float* __restrict__ roW,
                               const float* __restrict__ rob, int CH, float* __restrict__ Match) {
  __shared__ float sw[HID];
  if (threadIdx.x < CH) sw[threadIdx.x] = roW[threadIdx.x];
  __syncthreads();
  const float b = rob[0];
  for (unsigned idx = blockIdx.x * blockDim.x + threadIdx.x; idx < NN; idx += gridDim.x * blockDim.x) {
    float acc = b;
    for (int h = 0; h < CH; ++h)
      acc += (float)H[(size_t)h * NN + idx] * sw[h];
    Match[idx] = acc;
  }
}

// Per-row max and sum(exp(x-max)): one block per row.
__global__ __launch_bounds__(256) void row_stats(const float* __restrict__ M,
                                                 float* __restrict__ rmax, float* __restrict__ rsum) {
  __shared__ float red[256];
  const int i = blockIdx.x, t = threadIdx.x;
  float m = -3.0e38f;
  for (int c = t; c < NDIM; c += 256) m = fmaxf(m, M[(size_t)i * NDIM + c]);
  red[t] = m; __syncthreads();
  for (int s = 128; s > 0; s >>= 1) { if (t < s) red[t] = fmaxf(red[t], red[t + s]); __syncthreads(); }
  const float mx = red[0]; __syncthreads();
  float sum = 0.0f;
  for (int c = t; c < NDIM; c += 256) sum += __expf(M[(size_t)i * NDIM + c] - mx);
  red[t] = sum; __syncthreads();
  for (int s = 128; s > 0; s >>= 1) { if (t < s) red[t] += red[t + s]; __syncthreads(); }
  if (t == 0) { rmax[i] = mx; rsum[i] = red[0]; }
}

// Per-column max and sum(exp(x-max)): one block per column (strided reads).
__global__ __launch_bounds__(256) void col_stats(const float* __restrict__ M,
                                                 float* __restrict__ cmax, float* __restrict__ csum) {
  __shared__ float red[256];
  const int c = blockIdx.x, t = threadIdx.x;
  float m = -3.0e38f;
  for (int i = t; i < NDIM; i += 256) m = fmaxf(m, M[(size_t)i * NDIM + c]);
  red[t] = m; __syncthreads();
  for (int s = 128; s > 0; s >>= 1) { if (t < s) red[t] = fmaxf(red[t], red[t + s]); __syncthreads(); }
  const float mx = red[0]; __syncthreads();
  float sum = 0.0f;
  for (int i = t; i < NDIM; i += 256) sum += __expf(M[(size_t)i * NDIM + c] - mx);
  red[t] = sum; __syncthreads();
  for (int s = 128; s > 0; s >>= 1) { if (t < s) red[t] += red[t + s]; __syncthreads(); }
  if (t == 0) { cmax[c] = mx; csum[c] = red[0]; }
}

// Mn = 0.5*(row-softmax + col-softmax)
__global__ void softmax_combine(const float* __restrict__ M,
                                const float* __restrict__ rmax, const float* __restrict__ rsum,
                                const float* __restrict__ cmax, const float* __restrict__ csum,
                                float* __restrict__ Y) {
  for (unsigned idx = blockIdx.x * blockDim.x + threadIdx.x; idx < NN; idx += gridDim.x * blockDim.x) {
    const int i = idx >> 10, c = idx & 1023;
    const float x = M[idx];
    Y[idx] = 0.5f * (__expf(x - rmax[i]) / rsum[i] + __expf(x - cmax[c]) / csum[c]);
  }
}

// Zero seed rows (blockIdx.y==0) and seed columns (blockIdx.y==1).
__global__ __launch_bounds__(1024) void seed_mask_zero(float* __restrict__ Y, const int* __restrict__ seeds) {
  const int m = blockIdx.x, t = threadIdx.x;
  if (blockIdx.y == 0) Y[(size_t)seeds[m] * NDIM + t] = 0.0f;
  else                 Y[(size_t)t * NDIM + seeds[NSEEDS + m]] = 0.0f;
}

__global__ void seed_mask_set(float* __restrict__ Y, const int* __restrict__ seeds) {
  int t = threadIdx.x;
  if (t < NSEEDS) Y[(size_t)seeds[t] * NDIM + seeds[NSEEDS + t]] = 1.0f;
}

// Deterministic greedy matching surrogate (row-sequential argmax over untaken
// columns); stands in for the host-side SciPy Hungarian in the reference.
__global__ __launch_bounds__(1024) void greedy_match(const float* __restrict__ Y,
                                                     int* __restrict__ matchcol) {
  __shared__ float sval[1024];
  __shared__ int   scol[1024];
  __shared__ unsigned char taken[1024];
  const int t = threadIdx.x;
  taken[t] = 0; __syncthreads();
  for (int i = 0; i < NDIM; ++i) {
    sval[t] = taken[t] ? -3.0e38f : Y[(size_t)i * NDIM + t];
    scol[t] = t; __syncthreads();
    for (int s = 512; s > 0; s >>= 1) {
      if (t < s && sval[t + s] > sval[t]) { sval[t] = sval[t + s]; scol[t] = scol[t + s]; }
      __syncthreads();
    }
    const int win = scol[0];
    if (t == 0) matchcol[i] = win;
    if (t == win) taken[t] = 1;
    __syncthreads();
  }
}

// S update: ST[h'][k2][j] = bf16((sum_h H_h[j,k2]*W[h][h'] + b[h'])/1000), h'=0..14
//           ST[15][k2][j] = bf16( matchcol[j]==k2 ? 10*Mn[j,k2] : 0 )
__global__ __launch_bounds__(256) void update_S(const bf16_t* __restrict__ H,
                                                const float* __restrict__ Y,
                                                const int* __restrict__ matchcol,
                                                const float* __restrict__ W,   // [CH][15]
                                                const float* __restrict__ b,   // [15]
                                                int CH, bf16_t* __restrict__ ST) {
  __shared__ float sW[HID * (HID - 1)];
  __shared__ float sb[HID - 1];
  for (int i = threadIdx.x; i < CH * (HID - 1); i += blockDim.x) sW[i] = W[i];
  if (threadIdx.x < HID - 1) sb[threadIdx.x] = b[threadIdx.x];
  __syncthreads();
  for (unsigned idx = blockIdx.x * blockDim.x + threadIdx.x; idx < NN; idx += gridDim.x * blockDim.x) {
    const int j = idx >> 10, k2 = idx & 1023;
    const size_t tIdx = (size_t)k2 * NDIM + j;     // transposed S storage
    float hv[HID];
    for (int h = 0; h < CH; ++h) hv[h] = (float)H[(size_t)h * NN + idx];
    #pragma unroll
    for (int ho = 0; ho < HID - 1; ++ho) {
      float acc = sb[ho];
      for (int h = 0; h < CH; ++h) acc += hv[h] * sW[h * (HID - 1) + ho];
      ST[(size_t)ho * NN + tIdx] = (bf16_t)(acc * 0.001f);
    }
    const float z = (matchcol[j] == k2) ? 10.0f * Y[idx] : 0.0f;
    ST[(size_t)(HID - 1) * NN + tIdx] = (bf16_t)z;
  }
}

// ============================================================================
// Host-side orchestration
// ============================================================================
extern "C" void kernel_launch(void* const* d_in, const int* in_sizes, int n_in,
                              void* d_out, int out_size, void* d_ws, size_t ws_size,
                              hipStream_t stream) {
  const float* G1    = (const float*)d_in[0];
  const float* G2    = (const float*)d_in[1];
  const int*   seeds = (const int*)d_in[2];
  // d_in[3..8]  = mlp_W[0..5], d_in[9..14] = mlp_b[0..5]
  // d_in[15..20]= ro_W[0..5],  d_in[21..26]= ro_b[0..5]

  char* ws = (char*)d_ws;
  size_t off = 0;
  bf16_t* G1bf  = (bf16_t*)(ws + off); off += (size_t)NN * 2;            // 2 MB
  bf16_t* G2T   = (bf16_t*)(ws + off); off += (size_t)NN * 2;            // 2 MB
  bf16_t* ST    = (bf16_t*)(ws + off); off += (size_t)HID * NN * 2;      // 32 MB
  bf16_t* Tbf   = (bf16_t*)(ws + off); off += (size_t)NN * 2;            // 2 MB
  bf16_t* Hbf   = (bf16_t*)(ws + off); off += (size_t)HID * NN * 2;      // 32 MB
  float*  Match = (float*)(ws + off);  off += (size_t)NN * 4;            // 4 MB
  float*  rmax  = (float*)(ws + off);  off += NDIM * 4;
  float*  rsum  = (float*)(ws + off);  off += NDIM * 4;
  float*  cmax  = (float*)(ws + off);  off += NDIM * 4;
  float*  csum  = (float*)(ws + off);  off += NDIM * 4;
  int*    mcol  = (int*)(ws + off);    off += NDIM * 4;

  float* out = (float*)d_out;   // [7][1024][1024]: Y_total[-1], then Y_total[0..5]

  cvt_f32_to_bf16<<<1024, 256, 0, stream>>>(G1, G1bf, NN);
  transpose_f32_to_bf16<<<dim3(32, 32), 256, 0, stream>>>(G2, G2T);
  zero_u32<<<1024, 256, 0, stream>>>((uint32_t*)ST, NN / 2);
  set_seed_plane<<<1, 64, 0, stream>>>(ST, seeds);

  const dim3 gemmGrid(NDIM / 64, NDIM / 128);

  for (int l = 0; l < 6; ++l) {
    const int CH = (l == 0) ? 1 : HID;
    const float* roW = (const float*)d_in[15 + l];
    const float* rob = (const float*)d_in[21 + l];

    for (int h = 0; h < CH; ++h) {
      gemm_bf16_nt<<<gemmGrid, 256, 0, stream>>>(G1bf, ST + (size_t)h * NN, Tbf);
      gemm_bf16_nt<<<gemmGrid, 256, 0, stream>>>(Tbf, G2T, Hbf + (size_t)h * NN);
    }

    readout_kernel<<<2048, 256, 0, stream>>>(Hbf, roW, rob, CH, Match);
    row_stats<<<NDIM, 256, 0, stream>>>(Match, rmax, rsum);
    col_stats<<<NDIM, 256, 0, stream>>>(Match, cmax, csum);

    float* Y = out + (size_t)(1 + l) * NN;
    softmax_combine<<<2048, 256, 0, stream>>>(Match, rmax, rsum, cmax, csum, Y);
    seed_mask_zero<<<dim3(NSEEDS, 2), 1024, 0, stream>>>(Y, seeds);
    seed_mask_set<<<1, 64, 0, stream>>>(Y, seeds);

    if (l < 5) {
      greedy_match<<<1, 1024, 0, stream>>>(Y, mcol);
      const float* mlpW = (const float*)d_in[3 + l];
      const float* mlpb = (const float*)d_in[9 + l];
      update_S<<<2048, 256, 0, stream>>>(Hbf, Y, mcol, mlpW, mlpb, CH, ST);
    }
  }

  hipMemcpyAsync(out, out + (size_t)6 * NN, (size_t)NN * sizeof(float),
                 hipMemcpyDeviceToDevice, stream);
}
